// SpatialDiffAttention_44659069944191
// MI455X (gfx1250) — matrix operand order repaired and checked by reading, not compiled, not run on hardware
//
#include <hip/hip_runtime.h>
#include <math.h>

// ---------------------------------------------------------------------------
// SpatialDiffAttention for MI455X (gfx1250), wave32, WMMA bf16 path.
// B=2, C=224, H=W=56 (N=3136), 7 heads x dim 32.
// ---------------------------------------------------------------------------

#define Bsz 2
#define Cc  224
#define Hh  56
#define Ww  56
#define Nn  3136
#define NH  7
#define HD  32

typedef __attribute__((ext_vector_type(16))) __bf16 v16bf;
typedef __attribute__((ext_vector_type(8)))  float  v8f;

union BV { v16bf v; unsigned u[8]; };

__device__ __forceinline__ unsigned short f2bfu(float f) {
  unsigned u = __builtin_bit_cast(unsigned, f);
  u += 0x7FFFu + ((u >> 16) & 1u);           // round-to-nearest-even
  return (unsigned short)(u >> 16);
}
__device__ __forceinline__ unsigned packbf2(float lo, float hi) {
  return (unsigned)f2bfu(lo) | ((unsigned)f2bfu(hi) << 16);
}
__device__ __forceinline__ float sigmoidf_(float x) {
  return 1.0f / (1.0f + __expf(-x));
}
__device__ __forceinline__ v8f wmma_bf16(v16bf a, v16bf b, v8f c) {
  // (neg_a, A, neg_b, B, c_mod, C, reuse_a, reuse_b)
  return __builtin_amdgcn_wmma_f32_16x16x32_bf16(false, a, false, b,
                                                 (short)0, c, false, false);
}

// ---------------------------------------------------------------------------
// lam = clip(exp(lq1.lk1) - exp(lq2.lk2), 0, 1) + lambda_init
// ---------------------------------------------------------------------------
__global__ void lam_kernel(const float* __restrict__ lq1, const float* __restrict__ lk1,
                           const float* __restrict__ lq2, const float* __restrict__ lk2,
                           const float* __restrict__ lam_init, float* __restrict__ lam_out) {
  int i = threadIdx.x;                        // 32 threads
  float p1 = lq1[i] * lk1[i];
  float p2 = lq2[i] * lk2[i];
  #pragma unroll
  for (int m = 16; m >= 1; m >>= 1) {
    p1 += __shfl_xor(p1, m, 32);
    p2 += __shfl_xor(p2, m, 32);
  }
  if (i == 0) {
    float d = expf(p1) - expf(p2);
    d = fminf(fmaxf(d, 0.0f), 1.0f);
    lam_out[0] = d + lam_init[0];
  }
}

// ---------------------------------------------------------------------------
// gated = D * sigmoid(rowmean + colmean); per-channel instance norm.
// one block per (b,c)
// ---------------------------------------------------------------------------
__global__ __launch_bounds__(256) void gate_norm_kernel(
    const float* __restrict__ D, const float* __restrict__ gnw,
    const float* __restrict__ gnb, float* __restrict__ SE) {
  __shared__ float tile[Nn];
  __shared__ float rowm[Hh];
  __shared__ float colm[Ww];
  __shared__ float red[256], red2[256];
  int bc = blockIdx.x;
  int c  = bc % Cc;
  int tid = threadIdx.x;
  const float* Dg = D + (size_t)bc * Nn;

  for (int i = tid; i < Nn; i += 256) tile[i] = Dg[i];
  __syncthreads();

  if (tid < Hh) {
    float s = 0.f;
    for (int w = 0; w < Ww; ++w) s += tile[tid * Ww + w];
    rowm[tid] = s * (1.0f / Ww);
  } else if (tid >= 64 && tid < 64 + Ww) {
    int w = tid - 64;
    float s = 0.f;
    for (int h = 0; h < Hh; ++h) s += tile[h * Ww + w];
    colm[w] = s * (1.0f / Hh);
  }
  __syncthreads();

  float s = 0.f, s2 = 0.f;
  for (int i = tid; i < Nn; i += 256) {
    int h = i / Ww, w = i % Ww;
    float g = tile[i] * sigmoidf_(rowm[h] + colm[w]);
    tile[i] = g;
    s += g; s2 += g * g;
  }
  red[tid] = s; red2[tid] = s2;
  __syncthreads();
  for (int off = 128; off > 0; off >>= 1) {
    if (tid < off) { red[tid] += red[tid + off]; red2[tid] += red2[tid + off]; }
    __syncthreads();
  }
  float mu  = red[0] * (1.0f / Nn);
  float var = red2[0] * (1.0f / Nn) - mu * mu;
  float inv = rsqrtf(var + 1e-5f);
  float gw = gnw[c], gb = gnb[c];
  float* outp = SE + (size_t)bc * Nn;
  for (int i = tid; i < Nn; i += 256)
    outp[i] = (tile[i] - mu) * inv * gw + gb;
}

// ---------------------------------------------------------------------------
// 1x1 conv as GEMM: Y[b,o,n] = bias[o] + sum_c W[o,c] * X[b,c,n]
// grid (49, O/16, B), block 256; 16 o x 64 n per block.
// ---------------------------------------------------------------------------
__global__ __launch_bounds__(256) void proj_kernel(
    const float* __restrict__ W, const float* __restrict__ bias,
    const float* __restrict__ X, float* __restrict__ Y, int O) {
  __shared__ float Wsh[16 * Cc];
  int b = blockIdx.z;
  const float* Xb = X + (size_t)b * Cc * Nn;
  float* Yb = Y + (size_t)b * O * Nn;
  int o0 = blockIdx.y * 16;
  int tid = threadIdx.x;

  for (int i = tid; i < 16 * Cc; i += 256)
    Wsh[i] = W[(size_t)(o0 + i / Cc) * Cc + (i % Cc)];
  __syncthreads();

  int n = blockIdx.x * 64 + (tid & 63);
  int orow = tid >> 6;                         // 0..3
  float a0 = 0.f, a1 = 0.f, a2 = 0.f, a3 = 0.f;
  for (int c = 0; c < Cc; ++c) {
    float x = Xb[(size_t)c * Nn + n];
    a0 += Wsh[(orow + 0)  * Cc + c] * x;
    a1 += Wsh[(orow + 4)  * Cc + c] * x;
    a2 += Wsh[(orow + 8)  * Cc + c] * x;
    a3 += Wsh[(orow + 12) * Cc + c] * x;
  }
  Yb[(size_t)(o0 + orow)      * Nn + n] = a0 + bias[o0 + orow];
  Yb[(size_t)(o0 + orow + 4)  * Nn + n] = a1 + bias[o0 + orow + 4];
  Yb[(size_t)(o0 + orow + 8)  * Nn + n] = a2 + bias[o0 + orow + 8];
  Yb[(size_t)(o0 + orow + 12) * Nn + n] = a3 + bias[o0 + orow + 12];
}

// ---------------------------------------------------------------------------
// Flash-style differential attention. grid (49, 7, B), block 128 (4 waves).
// Each wave owns one 16-row query tile; sweeps keys 32 at a time.
// WMMA bf16 16x16x32, f32 accumulate. LDS rows padded to 34 (bank spread).
// Next key tile is prefetched (global_prefetch_b8) while current computes.
// ---------------------------------------------------------------------------
#define LPAD 34
__global__ __launch_bounds__(128) void attn_kernel(
    const float* __restrict__ Qp, const float* __restrict__ Kp,
    const float* __restrict__ Vp, const float* __restrict__ lamp,
    float* __restrict__ AO) {
  __shared__ unsigned short K0t[32 * LPAD];   // [m][d] bf16
  __shared__ unsigned short K1t[32 * LPAD];   // [m][d] bf16
  __shared__ unsigned short Vt [32 * LPAD];   // [d][m] bf16
  __shared__ unsigned short Ps [4][16 * LPAD];// per-wave [n][m] bf16

  const int b = blockIdx.z, h = blockIdx.y;
  const int wave = threadIdx.x >> 5, lane = threadIdx.x & 31;
  const int n0 = (blockIdx.x * 4 + wave) * 16;
  const float lam = lamp[0];
  const float scale = 0.17677669529663687f;   // 32^-0.5

  const float* Q0  = Qp + ((size_t)(b * 2 + 0) * Cc + h * HD) * Nn;
  const float* Q1  = Qp + ((size_t)(b * 2 + 1) * Cc + h * HD) * Nn;
  const float* K0g = Kp + ((size_t)(b * 2 + 0) * Cc + h * HD) * Nn;
  const float* K1g = Kp + ((size_t)(b * 2 + 1) * Cc + h * HD) * Nn;
  const float* Vg  = Vp + ((size_t)b * Cc + h * HD) * Nn;

  const int mrow  = lane & 15;
  const int half  = lane >> 4;
  const int kbase = half * 8;

  // A-tiles of Q0^T / Q1^T: [16 n x 32 d], loaded once.
  BV aQ0, aQ1;
  #pragma unroll
  for (int v = 0; v < 8; ++v) {
    int k = (v < 4) ? (kbase + 2 * v) : (16 + kbase + 2 * (v - 4));
    aQ0.u[v] = packbf2(Q0[(size_t)k * Nn + n0 + mrow],
                       Q0[(size_t)(k + 1) * Nn + n0 + mrow]);
    aQ1.u[v] = packbf2(Q1[(size_t)k * Nn + n0 + mrow],
                       Q1[(size_t)(k + 1) * Nn + n0 + mrow]);
  }

  v8f o0 = {0,0,0,0,0,0,0,0};
  v8f o1 = {0,0,0,0,0,0,0,0};
  float rm[8], lsum[8];
  #pragma unroll
  for (int v = 0; v < 8; ++v) { rm[v] = -1e30f; lsum[v] = 0.0f; }

  unsigned short* Pw = &Ps[wave][0];

  for (int m0 = 0; m0 < Nn; m0 += 32) {
    __syncthreads();
    // cooperative stage of K0/K1 (transposed) and V tiles, f32 -> bf16
    for (int i = threadIdx.x; i < 1024; i += 128) {
      int d = i >> 5, m = i & 31;
      K0t[m * LPAD + d] = f2bfu(K0g[(size_t)d * Nn + m0 + m]);
      K1t[m * LPAD + d] = f2bfu(K1g[(size_t)d * Nn + m0 + m]);
      Vt [d * LPAD + m] = f2bfu(Vg [(size_t)d * Nn + m0 + m]);
    }
    // prefetch next iteration's 96 rows into the cache hierarchy
    // (global_prefetch_b8: no counters touched, EXEC re-converges below)
    if (m0 + 32 < Nn) {
      int t = threadIdx.x;
      if (t < 32)       __builtin_prefetch(&K0g[(size_t)t * Nn + m0 + 32], 0, 0);
      else if (t < 64)  __builtin_prefetch(&K1g[(size_t)(t - 32) * Nn + m0 + 32], 0, 0);
      else if (t < 96)  __builtin_prefetch(&Vg [(size_t)(t - 64) * Nn + m0 + 32], 0, 0);
    }
    __syncthreads();

    // ---- S = scale*(Q0K0 - lam*Q1K1) for two 16-col tiles -------------
    v8f sa, sb;
    {
      v8f z = {0,0,0,0,0,0,0,0};
      BV bK0, bK1;
      #pragma unroll
      for (int v = 0; v < 8; ++v) {
        int addr = mrow * LPAD + ((v < 4) ? (kbase + 2 * v) : (16 + kbase + 2 * (v - 4)));        // m-half 0
        bK0.u[v] = *(const unsigned*)&K0t[addr];
        bK1.u[v] = *(const unsigned*)&K1t[addr];
      }
      v8f s1 = wmma_bf16(aQ0.v, bK0.v, z);
      v8f s2 = wmma_bf16(aQ1.v, bK1.v, z);
      #pragma unroll
      for (int e = 0; e < 8; ++e) sa[e] = scale * (s1[e] - lam * s2[e]);

      #pragma unroll
      for (int v = 0; v < 8; ++v) {
        int addr = (16 + mrow) * LPAD + ((v < 4) ? (kbase + 2 * v) : (16 + kbase + 2 * (v - 4))); // m-half 1
        bK0.u[v] = *(const unsigned*)&K0t[addr];
        bK1.u[v] = *(const unsigned*)&K1t[addr];
      }
      s1 = wmma_bf16(aQ0.v, bK0.v, z);
      s2 = wmma_bf16(aQ1.v, bK1.v, z);
      #pragma unroll
      for (int e = 0; e < 8; ++e) sb[e] = scale * (s1[e] - lam * s2[e]);
    }

    // ---- online softmax; P written to per-wave LDS (C->A relayout) ----
    #pragma unroll
    for (int v = 0; v < 8; ++v) {
      float mx = fmaxf(sa[v], sb[v]);
      mx = fmaxf(mx, __shfl_xor(mx, 1, 32));
      mx = fmaxf(mx, __shfl_xor(mx, 2, 32));
      mx = fmaxf(mx, __shfl_xor(mx, 4, 32));
      mx = fmaxf(mx, __shfl_xor(mx, 8, 32));
      float mnew  = fmaxf(rm[v], mx);
      float alpha = __expf(rm[v] - mnew);
      float pa = __expf(sa[v] - mnew);
      float pb = __expf(sb[v] - mnew);
      float rs = pa + pb;
      rs += __shfl_xor(rs, 1, 32);
      rs += __shfl_xor(rs, 2, 32);
      rs += __shfl_xor(rs, 4, 32);
      rs += __shfl_xor(rs, 8, 32);
      lsum[v] = lsum[v] * alpha + rs;
      rm[v]   = mnew;
      o0[v] *= alpha;
      o1[v] *= alpha;
      int nrow = v + 8 * half;
      Pw[nrow * LPAD + mrow]      = f2bfu(pa);
      Pw[nrow * LPAD + 16 + mrow] = f2bfu(pb);
    }
    __syncthreads();

    // ---- O += P @ V^T ------------------------------------------------
    BV aP, bV0, bV1;
    #pragma unroll
    for (int v = 0; v < 8; ++v) {
      int k = (v < 4) ? (kbase + 2 * v) : (16 + kbase + 2 * (v - 4));
      aP.u[v]  = *(const unsigned*)&Pw[mrow * LPAD + k];
      bV0.u[v] = *(const unsigned*)&Vt[mrow * LPAD + k];
      bV1.u[v] = *(const unsigned*)&Vt[(16 + mrow) * LPAD + k];
    }
    o0 = wmma_bf16(aP.v, bV0.v, o0);
    o1 = wmma_bf16(aP.v, bV1.v, o1);
  }

  // epilogue: divide by row sums, write out[b, h*32+d, n]
  #pragma unroll
  for (int v = 0; v < 8; ++v) {
    int nrow = v + 8 * half;
    float inv = 1.0f / lsum[v];
    AO[((size_t)b * Cc + h * HD + mrow)      * Nn + n0 + nrow] = o0[v] * inv;
    AO[((size_t)b * Cc + h * HD + 16 + mrow) * Nn + n0 + nrow] = o1[v] * inv;
  }
}

// ---------------------------------------------------------------------------
// x1w[b,c] = sigmoid(mean_n out[b,c,n]) * wo[c]
// ---------------------------------------------------------------------------
__global__ __launch_bounds__(256) void x1_kernel(
    const float* __restrict__ AO, const float* __restrict__ wo,
    float* __restrict__ x1w) {
  __shared__ float red[256];
  int bc = blockIdx.x, tid = threadIdx.x;
  const float* a = AO + (size_t)bc * Nn;
  float s = 0.f;
  for (int i = tid; i < Nn; i += 256) s += a[i];
  red[tid] = s; __syncthreads();
  for (int off = 128; off > 0; off >>= 1) {
    if (tid < off) red[tid] += red[tid + off];
    __syncthreads();
  }
  if (tid == 0) {
    int c = bc % Cc;
    x1w[bc] = sigmoidf_(red[0] * (1.0f / Nn)) * wo[c];
  }
}

// ---------------------------------------------------------------------------
// Fold per-channel gate + 1x1 head into an effective 3x3 filter:
// we[b,c',k] = sum_c x1w[b,c]*w3[c,c',k];  be[b] = bo + sum_c x1w[b,c]*b3[c]
// ---------------------------------------------------------------------------
__global__ __launch_bounds__(128) void wfold_kernel(
    const float* __restrict__ x1w, const float* __restrict__ w3,
    const float* __restrict__ b3, const float* __restrict__ bo,
    float* __restrict__ we, float* __restrict__ be) {
  int b = blockIdx.y;
  int t = blockIdx.x * 128 + threadIdx.x;
  if (t < Cc * 9) {
    int cp = t / 9, k = t % 9;
    float acc = 0.f;
    for (int c = 0; c < Cc; ++c)
      acc += x1w[b * Cc + c] * w3[((size_t)c * Cc + cp) * 9 + k];
    we[(size_t)b * Cc * 9 + t] = acc;
  }
  if (blockIdx.x == 0 && threadIdx.x == 0) {
    float acc = bo[0];
    for (int c = 0; c < Cc; ++c) acc += x1w[b * Cc + c] * b3[c];
    be[b] = acc;
  }
}

// ---------------------------------------------------------------------------
// attn_map[b,h,w] = sigmoid(be[b] + sum_{c',3x3} we[b,c',k]*out_pad[b,c',..])
// grid (49, B), block 64 (one pixel per thread)
// ---------------------------------------------------------------------------
__global__ __launch_bounds__(64) void final_kernel(
    const float* __restrict__ AO, const float* __restrict__ we,
    const float* __restrict__ be, float* __restrict__ out) {
  __shared__ float wsh[Cc * 9];
  int b = blockIdx.y, tid = threadIdx.x;
  for (int i = tid; i < Cc * 9; i += 64) wsh[i] = we[(size_t)b * Cc * 9 + i];
  __syncthreads();

  int p = blockIdx.x * 64 + tid;
  int h = p / Ww, w = p % Ww;
  float acc = be[b];
  for (int cp = 0; cp < Cc; ++cp) {
    const float* base = AO + ((size_t)b * Cc + cp) * Nn;
    const float* wk = &wsh[cp * 9];
    #pragma unroll
    for (int dy = 0; dy < 3; ++dy) {
      int hh = h + dy - 1;
      if (hh < 0 || hh >= Hh) continue;
      const float* row = base + hh * Ww;
      #pragma unroll
      for (int dx = 0; dx < 3; ++dx) {
        int ww = w + dx - 1;
        if (ww < 0 || ww >= Ww) continue;
        acc += wk[dy * 3 + dx] * row[ww];
      }
    }
  }
  out[(size_t)b * Nn + p] = sigmoidf_(acc);
}

// ---------------------------------------------------------------------------
extern "C" void kernel_launch(void* const* d_in, const int* in_sizes, int n_in,
                              void* d_out, int out_size, void* d_ws, size_t ws_size,
                              hipStream_t stream) {
  const float* D        = (const float*)d_in[0];
  const float* wq       = (const float*)d_in[1];
  const float* bq       = (const float*)d_in[2];
  const float* wk       = (const float*)d_in[3];
  const float* bk       = (const float*)d_in[4];
  const float* wv       = (const float*)d_in[5];
  const float* bv       = (const float*)d_in[6];
  const float* gnw      = (const float*)d_in[7];
  const float* gnb      = (const float*)d_in[8];
  const float* lam_init = (const float*)d_in[9];
  const float* lq1      = (const float*)d_in[10];
  const float* lk1      = (const float*)d_in[11];
  const float* lq2      = (const float*)d_in[12];
  const float* lk2      = (const float*)d_in[13];
  const float* w3       = (const float*)d_in[14];
  const float* b3       = (const float*)d_in[15];
  const float* wo       = (const float*)d_in[16];
  const float* bo       = (const float*)d_in[17];
  float* out = (float*)d_out;

  // workspace layout (floats): ~39.4 MB total
  float* ws  = (float*)d_ws;
  float* SE  = ws;                                   // B*C*N
  float* Q   = SE  + (size_t)Bsz * Cc * Nn;          // B*2C*N
  float* Kb  = Q   + (size_t)Bsz * 2 * Cc * Nn;      // B*2C*N
  float* V   = Kb  + (size_t)Bsz * 2 * Cc * Nn;      // B*C*N
  float* AO  = V   + (size_t)Bsz * Cc * Nn;          // B*C*N
  float* x1w = AO  + (size_t)Bsz * Cc * Nn;          // B*C
  float* we  = x1w + Bsz * Cc;                       // B*C*9
  float* be  = we  + Bsz * Cc * 9;                   // B
  float* lam = be  + Bsz;                            // 1

  lam_kernel<<<1, 32, 0, stream>>>(lq1, lk1, lq2, lk2, lam_init, lam);
  gate_norm_kernel<<<Bsz * Cc, 256, 0, stream>>>(D, gnw, gnb, SE);
  proj_kernel<<<dim3(49, 28, Bsz), 256, 0, stream>>>(wq, bq, SE, Q, 2 * Cc);
  proj_kernel<<<dim3(49, 28, Bsz), 256, 0, stream>>>(wk, bk, D,  Kb, 2 * Cc);
  proj_kernel<<<dim3(49, 14, Bsz), 256, 0, stream>>>(wv, bv, D,  V,  Cc);
  attn_kernel<<<dim3(49, NH, Bsz), 128, 0, stream>>>(Q, Kb, V, lam, AO);
  x1_kernel<<<Bsz * Cc, 256, 0, stream>>>(AO, wo, x1w);
  wfold_kernel<<<dim3(16, Bsz), 128, 0, stream>>>(x1w, w3, b3, bo, we, be);
  final_kernel<<<dim3(49, Bsz), 64, 0, stream>>>(AO, we, be, out);
}
